// MultiHeadAttentionPooling_42803644071951
// MI455X (gfx1250) — matrix operand order repaired and checked by reading, not compiled, not run on hardware
//
#include <hip/hip_runtime.h>

#define NTOK 4096
#define INDIM 512
#define NHEADS 8
#define HDIM 64
#define TOT 512
#define LN_EPS 1e-5f
#define LDS_LD 72  // 64 f16 + 8 f16 pad -> 144B row stride, spreads LDS banks

typedef __attribute__((ext_vector_type(16))) _Float16 v16h;
typedef __attribute__((ext_vector_type(8)))  _Float16 v8h;
typedef __attribute__((ext_vector_type(8)))  float    v8f;

// ---------------- WMMA fragment helpers (CDNA5 16x16x32 f16 layouts) ----------------
// A (16x32, row M = lane&15): lanes 0-15 hold K 0..15, lanes 16-31 hold K 16..31 (contiguous)
static __device__ __forceinline__ v16h load_a_frag(const _Float16* tile, int ld) {
  const int lane = threadIdx.x & 31;
  const _Float16* p = tile + (size_t)(lane & 15) * ld + ((lane >> 4) << 4);
  v8h lo = *(const v8h*)(p);
  v8h hi = *(const v8h*)(p + 8);
  v16h f;
#pragma unroll
  for (int i = 0; i < 8; ++i) { f[i] = lo[i]; f[i + 8] = hi[i]; }
  return f;
}

// B (32x16, col N = lane&15): lanes 0-15 hold K 0..15, lanes 16-31 hold K 16..31 (contiguous)
static __device__ __forceinline__ v16h load_b_frag(const _Float16* tile, int ld) {
  const int lane = threadIdx.x & 31;
  const _Float16* p = tile + (size_t)(lane & 15) * ld + ((lane >> 4) << 4);
  return *(const v16h*)(p);
}

// B fragment from LDS-staged tile (16B-granular loads: padded rows are only 16B aligned)
static __device__ __forceinline__ v16h load_b_frag_lds(const _Float16* tile, int ld) {
  const int lane = threadIdx.x & 31;
  const _Float16* p = tile + (lane & 15) * ld + ((lane >> 4) << 4);
  v8h lo = *(const v8h*)(p);
  v8h hi = *(const v8h*)(p + 8);
  v16h f;
#pragma unroll
  for (int i = 0; i < 8; ++i) { f[i] = lo[i]; f[i + 8] = hi[i]; }
  return f;
}

static __device__ __forceinline__ v8f wmma16(v16h a, v16h b, v8f c) {
  return __builtin_amdgcn_wmma_f32_16x16x32_f16(false, a, false, b, (short)0, c,
                                                false, false);
}

// async copy of one 16B chunk: global -> LDS (GV mode), tracked by ASYNCcnt
static __device__ __forceinline__ void async_copy16(unsigned lds_off, const void* gaddr) {
  asm volatile("global_load_async_to_lds_b128 %0, %1, off"
               :: "v"(lds_off), "v"(gaddr) : "memory");
}
static __device__ __forceinline__ void wait_async0() {
  asm volatile("s_wait_asynccnt 0x0" ::: "memory");
}

// ---------------- prep kernels ----------------
__global__ void k_cvt_h(const float* __restrict__ h, _Float16* __restrict__ hh) {
  int i = blockIdx.x * blockDim.x + threadIdx.x;
  hh[i] = (_Float16)h[i];
}

// WT[o*512 + i] = W[i*512 + o], fp32 -> f16
__global__ void k_transpose_w(const float* __restrict__ W, _Float16* __restrict__ WT) {
  int idx = blockIdx.x * blockDim.x + threadIdx.x;
  int o = idx >> 9, i = idx & 511;
  WT[idx] = (_Float16)W[(size_t)i * TOT + o];
}

// ---------------- projection GEMM: Out = (hh @ W + b) * scale, f16 out ----------------
__global__ __launch_bounds__(128) void k_proj(const _Float16* __restrict__ hh,
                                              const _Float16* __restrict__ WT,
                                              const float* __restrict__ bias,
                                              _Float16* __restrict__ outm,
                                              float scale) {
  const int wave = threadIdx.x >> 5;
  const int lane = threadIdx.x & 31;
  const int m0 = (blockIdx.x * 4 + wave) * 16;
  const int n0 = blockIdx.y * 64;
  v8f c0 = {}, c1 = {}, c2 = {}, c3 = {};
  for (int k0 = 0; k0 < INDIM; k0 += 32) {
    v16h a  = load_a_frag(hh + (size_t)m0 * INDIM + k0, INDIM);
    v16h b0 = load_b_frag(WT + (size_t)(n0 +  0) * INDIM + k0, INDIM);
    v16h b1 = load_b_frag(WT + (size_t)(n0 + 16) * INDIM + k0, INDIM);
    v16h b2 = load_b_frag(WT + (size_t)(n0 + 32) * INDIM + k0, INDIM);
    v16h b3 = load_b_frag(WT + (size_t)(n0 + 48) * INDIM + k0, INDIM);
    c0 = wmma16(a, b0, c0);
    c1 = wmma16(a, b1, c1);
    c2 = wmma16(a, b2, c2);
    c3 = wmma16(a, b3, c3);
  }
  const int ncol = lane & 15;
  const int mb = (lane >> 4) << 3;  // C/D: lanes>=16 hold M = r+8
  v8f cc[4] = {c0, c1, c2, c3};
#pragma unroll
  for (int t = 0; t < 4; ++t) {
    int col = n0 + t * 16 + ncol;
    float b = bias[col];
#pragma unroll
    for (int r = 0; r < 8; ++r)
      outm[(size_t)(m0 + mb + r) * TOT + col] = (_Float16)((cc[t][r] + b) * scale);
  }
}

// ---------------- pass 1: inv row-sums of exp2(scores) ----------------
// grid (64, NHEADS), block 128. Wave owns 16 query rows; block streams key tiles
// through a double-buffered async-LDS stage shared by all 4 waves.
__global__ __launch_bounds__(128) void k_rowsum(const _Float16* __restrict__ Qh,
                                                const _Float16* __restrict__ Kh,
                                                float* __restrict__ invrs) {
  __shared__ _Float16 sbuf[2][16][LDS_LD];
  const int tid  = threadIdx.x;
  const int wave = tid >> 5;
  const int lane = tid & 31;
  const int hd = blockIdx.y;
  const int n0 = (blockIdx.x * 4 + wave) * 16;
  const _Float16* qb = Qh + (size_t)n0 * TOT + hd * HDIM;
  v16h a0 = load_a_frag(qb, TOT);
  v16h a1 = load_a_frag(qb + 32, TOT);

  // cooperative async-copy mapping: tile = 16 rows x 64 f16 = 128 chunks of 16B
  const int crow = tid >> 3;                 // 0..15
  const int ccol = (tid & 7) << 3;           // 0..56 step 8 (f16 elements)
  const _Float16* gsrc = Kh + hd * HDIM + (size_t)crow * TOT + ccol;
  unsigned lds_off[2];
  lds_off[0] = (unsigned)(size_t)&sbuf[0][crow][ccol];
  lds_off[1] = (unsigned)(size_t)&sbuf[1][crow][ccol];

  async_copy16(lds_off[0], gsrc);            // prologue: tile m0 = 0 -> buf 0

  float acc[8] = {0.f, 0.f, 0.f, 0.f, 0.f, 0.f, 0.f, 0.f};
  int cur = 0;
  for (int m0 = 0; m0 < NTOK; m0 += 16) {
    wait_async0();                           // my chunk of buf[cur] landed
    __syncthreads();                         // everyone's chunk landed
    if (m0 + 16 < NTOK)                      // prefetch next tile into other buffer
      async_copy16(lds_off[cur ^ 1], gsrc + (size_t)(m0 + 16) * TOT);
    v16h b0 = load_b_frag_lds(&sbuf[cur][0][0],  LDS_LD);
    v16h b1 = load_b_frag_lds(&sbuf[cur][0][32], LDS_LD);
    v8f c = {};
    c = wmma16(a0, b0, c);
    c = wmma16(a1, b1, c);
#pragma unroll
    for (int r = 0; r < 8; ++r) acc[r] += exp2f(c[r]);
    cur ^= 1;
  }
#pragma unroll
  for (int r = 0; r < 8; ++r) {
    float v = acc[r];
    v += __shfl_xor(v, 1, 32);
    v += __shfl_xor(v, 2, 32);
    v += __shfl_xor(v, 4, 32);
    v += __shfl_xor(v, 8, 32);
    acc[r] = v;
  }
  if ((lane & 15) == 0) {
    const int mb = (lane >> 4) << 3;
#pragma unroll
    for (int r = 0; r < 8; ++r)
      invrs[hd * NTOK + n0 + mb + r] = 1.0f / acc[r];
  }
}

// ---------------- pass 2: column weights w[h,m] = (1/N) * sum_n softmax(s)_nm --------
// grid (64, NHEADS), block 128. Wave owns 16 key rows (A); block streams query tiles.
__global__ __launch_bounds__(128) void k_colw(const _Float16* __restrict__ Kh,
                                              const _Float16* __restrict__ Qh,
                                              const float* __restrict__ invrs,
                                              float* __restrict__ wcol) {
  __shared__ _Float16 sbuf[2][16][LDS_LD];
  const int tid  = threadIdx.x;
  const int wave = tid >> 5;
  const int lane = tid & 31;
  const int hd = blockIdx.y;
  const int m0 = (blockIdx.x * 4 + wave) * 16;
  const _Float16* kb = Kh + (size_t)m0 * TOT + hd * HDIM;
  v16h a0 = load_a_frag(kb, TOT);
  v16h a1 = load_a_frag(kb + 32, TOT);

  const int crow = tid >> 3;
  const int ccol = (tid & 7) << 3;
  const _Float16* gsrc = Qh + hd * HDIM + (size_t)crow * TOT + ccol;
  unsigned lds_off[2];
  lds_off[0] = (unsigned)(size_t)&sbuf[0][crow][ccol];
  lds_off[1] = (unsigned)(size_t)&sbuf[1][crow][ccol];

  async_copy16(lds_off[0], gsrc);

  float acc[8] = {0.f, 0.f, 0.f, 0.f, 0.f, 0.f, 0.f, 0.f};
  const float* irsb = invrs + hd * NTOK;
  int cur = 0;
  for (int n0 = 0; n0 < NTOK; n0 += 16) {
    wait_async0();
    __syncthreads();
    if (n0 + 16 < NTOK)
      async_copy16(lds_off[cur ^ 1], gsrc + (size_t)(n0 + 16) * TOT);
    v16h b0 = load_b_frag_lds(&sbuf[cur][0][0],  LDS_LD);
    v16h b1 = load_b_frag_lds(&sbuf[cur][0][32], LDS_LD);
    v8f c = {};
    c = wmma16(a0, b0, c);
    c = wmma16(a1, b1, c);
    float irs = irsb[n0 + (lane & 15)];  // this lane's query column
#pragma unroll
    for (int r = 0; r < 8; ++r) acc[r] += exp2f(c[r]) * irs;
    cur ^= 1;
  }
#pragma unroll
  for (int r = 0; r < 8; ++r) {
    float v = acc[r];
    v += __shfl_xor(v, 1, 32);
    v += __shfl_xor(v, 2, 32);
    v += __shfl_xor(v, 4, 32);
    v += __shfl_xor(v, 8, 32);
    acc[r] = v;
  }
  if ((lane & 15) == 0) {
    const int mb = (lane >> 4) << 3;
#pragma unroll
    for (int r = 0; r < 8; ++r)
      wcol[hd * NTOK + m0 + mb + r] = acc[r] * (1.0f / NTOK);
  }
}

// ---------------- fused tail: u[h,c] = sum_m w[h,m] h[m,c] and hmean, one pass over h
__global__ void k_uh(const float* __restrict__ h, const float* __restrict__ wcol,
                     float* __restrict__ uvec, float* __restrict__ hmean) {
  const int c = blockIdx.x * blockDim.x + threadIdx.x;  // 0..511
  float s[NHEADS] = {0.f, 0.f, 0.f, 0.f, 0.f, 0.f, 0.f, 0.f};
  float sm = 0.f;
  for (int m = 0; m < NTOK; ++m) {
    float hv = h[(size_t)m * INDIM + c];
    sm += hv;
#pragma unroll
    for (int hd = 0; hd < NHEADS; ++hd)
      s[hd] += wcol[hd * NTOK + m] * hv;   // block-uniform -> scalar loads
  }
#pragma unroll
  for (int hd = 0; hd < NHEADS; ++hd) uvec[hd * INDIM + c] = s[hd];
  hmean[c] = sm * (1.0f / NTOK);
}

// pooled = u @ Wv + bv ; fused = relu(pooled @ Wf + bf); x = fused + hmean; LayerNorm.
__global__ __launch_bounds__(512) void k_final(const float* __restrict__ uvec,
    const float* __restrict__ Wv, const float* __restrict__ bv,
    const float* __restrict__ Wf, const float* __restrict__ bf,
    const float* __restrict__ hmean, const float* __restrict__ gam,
    const float* __restrict__ bet, float* __restrict__ out) {
  __shared__ float pooled[TOT];
  __shared__ float sred[16];
  const int t = threadIdx.x;
  const int lane = t & 31;
  const int wv = t >> 5;
  const int hd = t >> 6;
  const float* ur = uvec + hd * INDIM;
  float s = 0.f;
  for (int c2 = 0; c2 < INDIM; ++c2) s += ur[c2] * Wv[(size_t)c2 * TOT + t];
  pooled[t] = s + bv[t];
  __syncthreads();
  float f = 0.f;
  for (int j = 0; j < TOT; ++j) f += pooled[j] * Wf[(size_t)j * INDIM + t];
  f += bf[t];
  f = fmaxf(f, 0.f);
  float x = f + hmean[t];

  // mean
  float v = x;
#pragma unroll
  for (int m = 1; m <= 16; m <<= 1) v += __shfl_xor(v, m, 32);
  if (lane == 0) sred[wv] = v;
  __syncthreads();
  if (t < 32) {
    float r = (t < 16) ? sred[t] : 0.0f;
#pragma unroll
    for (int m = 1; m <= 8; m <<= 1) r += __shfl_xor(r, m, 32);
    if (t == 0) sred[0] = r;
  }
  __syncthreads();
  float mu = sred[0] * (1.0f / TOT);
  __syncthreads();

  // variance
  float d = x - mu;
  v = d * d;
#pragma unroll
  for (int m = 1; m <= 16; m <<= 1) v += __shfl_xor(v, m, 32);
  if (lane == 0) sred[wv] = v;
  __syncthreads();
  if (t < 32) {
    float r = (t < 16) ? sred[t] : 0.0f;
#pragma unroll
    for (int m = 1; m <= 8; m <<= 1) r += __shfl_xor(r, m, 32);
    if (t == 0) sred[0] = r;
  }
  __syncthreads();
  float var = sred[0] * (1.0f / TOT);
  out[t] = d * rsqrtf(var + LN_EPS) * gam[t] + bet[t];
}

// ---------------- host launcher ----------------
extern "C" void kernel_launch(void* const* d_in, const int* in_sizes, int n_in,
                              void* d_out, int out_size, void* d_ws, size_t ws_size,
                              hipStream_t stream) {
  (void)in_sizes; (void)n_in; (void)out_size; (void)ws_size;
  const float* h   = (const float*)d_in[0];
  const float* Wq  = (const float*)d_in[1];
  const float* bq  = (const float*)d_in[2];
  const float* Wk  = (const float*)d_in[3];
  const float* bk  = (const float*)d_in[4];
  const float* Wv  = (const float*)d_in[5];
  const float* bv  = (const float*)d_in[6];
  const float* Wf  = (const float*)d_in[7];
  const float* bf  = (const float*)d_in[8];
  const float* gam = (const float*)d_in[9];
  const float* bet = (const float*)d_in[10];
  float* out = (float*)d_out;

  char* base = (char*)d_ws;
  size_t off = 0;
  auto carve = [&](size_t bytes) -> void* {
    void* p = base + off;
    off += (bytes + 255) & ~(size_t)255;
    return p;
  };
  _Float16* hh   = (_Float16*)carve((size_t)NTOK * INDIM * 2);
  _Float16* WqT  = (_Float16*)carve((size_t)INDIM * TOT * 2);
  _Float16* WkT  = (_Float16*)carve((size_t)INDIM * TOT * 2);
  _Float16* Qh   = (_Float16*)carve((size_t)NTOK * TOT * 2);
  _Float16* Kh   = (_Float16*)carve((size_t)NTOK * TOT * 2);
  float*    invrs = (float*)carve((size_t)NHEADS * NTOK * 4);
  float*    wcol  = (float*)carve((size_t)NHEADS * NTOK * 4);
  float*    hmean = (float*)carve((size_t)INDIM * 4);
  float*    uvec  = (float*)carve((size_t)NHEADS * INDIM * 4);

  const float LOG2E = 1.4426950408889634f;

  k_cvt_h<<<(NTOK * INDIM) / 256, 256, 0, stream>>>(h, hh);
  k_transpose_w<<<(INDIM * TOT) / 256, 256, 0, stream>>>(Wq, WqT);
  k_transpose_w<<<(INDIM * TOT) / 256, 256, 0, stream>>>(Wk, WkT);
  // Q pre-scaled by log2(e)/sqrt(HEAD_DIM) so softmax uses exp2 (v_exp_f32) directly.
  k_proj<<<dim3(64, 8), 128, 0, stream>>>(hh, WqT, bq, Qh, 0.125f * LOG2E);
  k_proj<<<dim3(64, 8), 128, 0, stream>>>(hh, WkT, bk, Kh, 1.0f);
  k_rowsum<<<dim3(64, NHEADS), 128, 0, stream>>>(Qh, Kh, invrs);
  k_colw<<<dim3(64, NHEADS), 128, 0, stream>>>(Kh, Qh, invrs, wcol);
  k_uh<<<2, 256, 0, stream>>>(h, wcol, uvec, hmean);
  k_final<<<1, 512, 0, stream>>>(uvec, Wv, bv, Wf, bf, hmean, gam, bet, out);
}
